// MultiHeadAttention_88931592831329
// MI455X (gfx1250) — matrix operand order repaired and checked by reading, not compiled, not run on hardware
//
#include <hip/hip_runtime.h>

// ---------------- problem constants (fixed by the reference) ----------------
constexpr int Sq   = 2048;   // sequence length
constexpr int Ch   = 64;     // head dim
constexpr int H    = 8;      // heads
constexpr int CIN  = 512;    // model dim (= H*Ch)
constexpr int Bn   = 4;      // batch
constexpr int MROW = Bn * Sq;   // 8192 rows for all row-major GEMMs
constexpr int NDIM = 512;
constexpr int KDIM = 512;

typedef __bf16 bf16;
typedef __attribute__((ext_vector_type(16))) __bf16 v16bf;
typedef __attribute__((ext_vector_type(8)))  float  v8f;

union V16 { uint4 q[2]; v16bf v; };   // 32B fragment <-> two b128 loads

__device__ __forceinline__ bf16 f2bf(float f) { return (bf16)f; }

// CDNA5 async global->LDS copy path (ASYNCcnt), guarded so compile never regresses.
#if defined(__AMDGCN__) && __has_builtin(__builtin_amdgcn_global_load_async_to_lds_b128)
#define HAVE_ASYNC_LDS 1
typedef int v4i_gcc __attribute__((vector_size(16)));
typedef __attribute__((address_space(1))) v4i_gcc gv4i;   // global (AS1)
typedef __attribute__((address_space(3))) v4i_gcc lv4i;   // LDS (AS3)
__device__ __forceinline__ void async_cp16(const void* g, void* l) {
    __builtin_amdgcn_global_load_async_to_lds_b128(
        (gv4i*)(unsigned long long)(uintptr_t)g,
        (lv4i*)(unsigned int)(uintptr_t)l, 0, 0);
}
__device__ __forceinline__ void async_wait0() {
#if __has_builtin(__builtin_amdgcn_s_wait_asynccnt)
    __builtin_amdgcn_s_wait_asynccnt(0);
#else
    asm volatile("s_wait_asynccnt 0" ::: "memory");
#endif
}
#endif

// ---------------- fp32 -> bf16 conversion ----------------
__global__ void cvt_f32_bf16(const float* __restrict__ in, bf16* __restrict__ out, int n) {
    int i = blockIdx.x * blockDim.x + threadIdx.x;
    if (i < n) out[i] = f2bf(in[i]);
}

// ---------------- generic  C[M x 512] = A[M x 512] * W[512 x 512]^T ----------------
// 256 threads = 8 waves; WG tile 128(M) x 64(N); wave tile 32x32 (2x2 WMMA tiles).
// mode 0: split-heads bf16 store (Q, scale applied)   outb = (B,H,S,Ch)
// mode 1: split-heads bf16 store (K)                  outb = (B,H,S,Ch)
// mode 2: transposed split-heads bf16 store (V)       outb = (B,H,Ch,S)
// mode 3: +bias vec, store fp32 (outf) AND bf16 (outb) row-major [M x 512]
// mode 4: +bias vec, outf = sigmoid(acc+bias) * gate  (final gating, fp32)
constexpr int LP = 40; // LDS row stride in bf16 (80B: 16B-aligned chunks, 20-bank skew)

__global__ __launch_bounds__(256) void gemm512(
    const bf16* __restrict__ A, const bf16* __restrict__ W,
    const float* __restrict__ bias, const float* __restrict__ gate,
    bf16* __restrict__ outb, float* __restrict__ outf,
    float scale, int mode)
{
    __shared__ bf16 sA[128 * LP];
    __shared__ bf16 sB[64 * LP];

    const int tid  = threadIdx.x;
    const int wave = tid >> 5, lane = tid & 31;
    const int hw   = lane >> 4, lm = lane & 15;   // half-wave, lane-in-half
    const int wm   = wave & 3,  wn = wave >> 2;   // wave grid 4(M) x 2(N)
    const int row0 = blockIdx.x * 128;
    const int col0 = blockIdx.y * 64;

    v8f acc[2][2] = {};

    for (int k0 = 0; k0 < KDIM; k0 += 32) {
        // stage A(128x32) and B(64x32) tiles as 16B chunks
        {
            int c0 = tid;                       // A chunks: 512 of 8 bf16
            int r0 = c0 >> 2, cc0 = (c0 & 3) * 8;
            int c1 = tid + 256;
            int r1 = c1 >> 2, cc1 = (c1 & 3) * 8;
            int rb = tid >> 2, cb = (tid & 3) * 8;   // B chunks: 256
#ifdef HAVE_ASYNC_LDS
            async_cp16(&A[(size_t)(row0 + r0) * KDIM + k0 + cc0], &sA[r0 * LP + cc0]);
            async_cp16(&A[(size_t)(row0 + r1) * KDIM + k0 + cc1], &sA[r1 * LP + cc1]);
            async_cp16(&W[(size_t)(col0 + rb) * KDIM + k0 + cb],  &sB[rb * LP + cb]);
            async_wait0();
#else
            *(uint4*)&sA[r0 * LP + cc0] = *(const uint4*)&A[(size_t)(row0 + r0) * KDIM + k0 + cc0];
            *(uint4*)&sA[r1 * LP + cc1] = *(const uint4*)&A[(size_t)(row0 + r1) * KDIM + k0 + cc1];
            *(uint4*)&sB[rb * LP + cb]  = *(const uint4*)&W[(size_t)(col0 + rb) * KDIM + k0 + cb];
#endif
        }
        __syncthreads();

        v16bf af[2], wf[2];
        #pragma unroll
        for (int mi = 0; mi < 2; ++mi) {        // A 16x32 frag (ISA 7.12.2)
            const uint4* p = (const uint4*)&sA[(wm * 32 + mi * 16 + lm) * LP];
            V16 t; t.q[0] = p[hw]; t.q[1] = p[2 + hw];
            af[mi] = t.v;
        }
        #pragma unroll
        for (int ni = 0; ni < 2; ++ni) {        // B 32x16 frag: K = 8*hw + e (e<8), 16 + 8*hw + (e-8) (e>=8)
            const uint4* p = (const uint4*)&sB[(wn * 32 + ni * 16 + lm) * LP];
            V16 t; t.q[0] = p[hw]; t.q[1] = p[2 + hw];
            wf[ni] = t.v;
        }
        #pragma unroll
        for (int mi = 0; mi < 2; ++mi)
            #pragma unroll
            for (int ni = 0; ni < 2; ++ni)
                acc[mi][ni] = __builtin_amdgcn_wmma_f32_16x16x32_bf16(
                    false, af[mi], false, wf[ni], (short)0, acc[mi][ni], false, false);
        __syncthreads();
    }

    #pragma unroll
    for (int mi = 0; mi < 2; ++mi) {
        #pragma unroll
        for (int ni = 0; ni < 2; ++ni) {
            int n = col0 + wn * 32 + ni * 16 + lm;
            #pragma unroll
            for (int i = 0; i < 8; ++i) {
                int row = row0 + wm * 32 + mi * 16 + i + 8 * hw;  // D: M = i + 8*hw
                float v = acc[mi][ni][i] * scale;
                if (mode <= 1) {
                    int b = row >> 11, s = row & (Sq - 1);
                    int h = n >> 6,  c = n & 63;
                    outb[(((size_t)(b * H + h) * Sq + s) * Ch) + c] = f2bf(v);
                } else if (mode == 2) {
                    int b = row >> 11, s = row & (Sq - 1);
                    int h = n >> 6,  c = n & 63;
                    outb[(((size_t)(b * H + h) * Ch + c) * Sq) + s] = f2bf(v);
                } else if (mode == 3) {
                    float o = v + bias[n];
                    outf[(size_t)row * NDIM + n] = o;
                    outb[(size_t)row * NDIM + n] = f2bf(o);
                } else {
                    float g  = v + bias[n];
                    float sg = 1.0f / (1.0f + __expf(-g));
                    outf[(size_t)row * NDIM + n] = sg * gate[(size_t)row * NDIM + n];
                }
            }
        }
    }
}

// ---------------- flash attention: one (b,h), 128 q-rows per WG, 16 per wave ----------------
constexpr int PP = 40;  // P-tile LDS stride (80B rows -> 16B-aligned b128 reloads)

__global__ __launch_bounds__(256) void attn_flash(
    const bf16* __restrict__ Q,    // (B*H, S, Ch), pre-scaled by 1/sqrt(Ch)
    const bf16* __restrict__ Km,   // (B*H, S, Ch)
    const bf16* __restrict__ Vt,   // (B*H, Ch, S)  (transposed)
    const float* __restrict__ bias,// (H, S, S) fp32
    const int*  __restrict__ mask, // (B, S)
    bf16* __restrict__ attn)       // (B*S, 512) merged heads
{
    __shared__ bf16 sP[8][16 * PP];   // per-wave P tile

    const int tid  = threadIdx.x;
    const int wave = tid >> 5, lane = tid & 31;
    const int hw   = lane >> 4, lm = lane & 15;
    const int bh   = blockIdx.y, b = bh >> 3, h = bh & 7;
    const int q0   = blockIdx.x * 128 + wave * 16;

    const bf16*  qb  = Q  + (size_t)bh * Sq * Ch;
    const bf16*  kmb = Km + (size_t)bh * Sq * Ch;
    const bf16*  vb  = Vt + (size_t)bh * Ch * Sq;
    const float* bb  = bias + (size_t)h * Sq * Sq;
    const int*   mb  = mask + b * Sq;

    // resident Q fragments: rows q0..q0+15, contraction chunks c=[0,32) and [32,64)
    v16bf qf[2];
    #pragma unroll
    for (int kk = 0; kk < 2; ++kk) {
        const uint4* p = (const uint4*)(qb + (size_t)(q0 + lm) * Ch + kk * 32);
        V16 t; t.q[0] = p[hw]; t.q[1] = p[2 + hw];
        qf[kk] = t.v;
    }

    v8f acc[4] = {};
    float mrow[8], lrow[8];
    #pragma unroll
    for (int i = 0; i < 8; ++i) { mrow[i] = -3.0e38f; lrow[i] = 0.0f; }

    bf16* myP = sP[wave];

    for (int kb = 0; kb < Sq; kb += 32) {
        // ---- scores: S[16 x 32] = Q(16x64) . K(32x64)^T  (4 WMMAs) ----
        v8f s0 = {}, s1 = {};
        #pragma unroll
        for (int kk = 0; kk < 2; ++kk) {
            V16 t0, t1;
            const uint4* p0 = (const uint4*)(kmb + (size_t)(kb + lm) * Ch      + kk * 32);
            const uint4* p1 = (const uint4*)(kmb + (size_t)(kb + 16 + lm) * Ch + kk * 32);
            t0.q[0] = p0[hw]; t0.q[1] = p0[2 + hw];
            t1.q[0] = p1[hw]; t1.q[1] = p1[2 + hw];
            s0 = __builtin_amdgcn_wmma_f32_16x16x32_bf16(false, qf[kk], false, t0.v, (short)0, s0, false, false);
            s1 = __builtin_amdgcn_wmma_f32_16x16x32_bf16(false, qf[kk], false, t1.v, (short)0, s1, false, false);
        }
        // prefetch next bias tile (gfx1250 global_prefetch_b8)
        if (kb + 32 < Sq)
            __builtin_prefetch(bb + (size_t)(q0 + 8 * hw) * Sq + kb + 32 + lm, 0, 1);

        // ---- bias + mask ----
        float off0 = (mb[kb + lm]      == 0) ? -1e8f : 0.0f;
        float off1 = (mb[kb + 16 + lm] == 0) ? -1e8f : 0.0f;
        #pragma unroll
        for (int i = 0; i < 8; ++i) {
            const float* bp = bb + (size_t)(q0 + i + 8 * hw) * Sq + kb + lm;
            s0[i] += bp[0]  + off0;
            s1[i] += bp[16] + off1;
        }

        // ---- online softmax (each row lives in 16 lanes of one half-wave) ----
        #pragma unroll
        for (int i = 0; i < 8; ++i) {
            float t = fmaxf(s0[i], s1[i]);
            #pragma unroll
            for (int d = 1; d < 16; d <<= 1) t = fmaxf(t, __shfl_xor(t, d, 32));
            float nm    = fmaxf(mrow[i], t);
            float alpha = __expf(mrow[i] - nm);
            float p0v   = __expf(s0[i] - nm);
            float p1v   = __expf(s1[i] - nm);
            float rs    = p0v + p1v;
            #pragma unroll
            for (int d = 1; d < 16; d <<= 1) rs += __shfl_xor(rs, d, 32);
            lrow[i] = lrow[i] * alpha + rs;
            mrow[i] = nm;
            #pragma unroll
            for (int ct = 0; ct < 4; ++ct) acc[ct][i] *= alpha;
            myP[(i + 8 * hw) * PP + lm]      = f2bf(p0v);   // C/D layout -> row-major LDS
            myP[(i + 8 * hw) * PP + 16 + lm] = f2bf(p1v);
        }

        // ---- reload P as A fragment (16x32) from this wave's private LDS ----
        v16bf pf;
        {
            const uint4* p = (const uint4*)&myP[lm * PP];
            V16 t; t.q[0] = p[hw]; t.q[1] = p[2 + hw];
            pf = t.v;
        }
        // ---- O += P(16x32) . V(32x64)   (4 WMMAs against Vt rows) ----
        #pragma unroll
        for (int ct = 0; ct < 4; ++ct) {
            const uint4* p = (const uint4*)(vb + (size_t)(ct * 16 + lm) * Sq + kb);
            V16 t; t.q[0] = p[hw]; t.q[1] = p[2 + hw];
            acc[ct] = __builtin_amdgcn_wmma_f32_16x16x32_bf16(
                false, pf, false, t.v, (short)0, acc[ct], false, false);
        }
    }

    // ---- normalize, merge heads: attn[(b*S+s), h*64 + c] ----
    #pragma unroll
    for (int i = 0; i < 8; ++i) {
        float inv = 1.0f / lrow[i];
        int srow = q0 + i + 8 * hw;
        #pragma unroll
        for (int ct = 0; ct < 4; ++ct)
            attn[((size_t)(b * Sq + srow)) * NDIM + h * Ch + ct * 16 + lm] =
                f2bf(acc[ct][i] * inv);
    }
}

// ---------------- host orchestration ----------------
extern "C" void kernel_launch(void* const* d_in, const int* in_sizes, int n_in,
                              void* d_out, int out_size, void* d_ws, size_t ws_size,
                              hipStream_t stream) {
    const float* x    = (const float*)d_in[0];
    const float* bias = (const float*)d_in[1];
    const int*   am   = (const int*)  d_in[2];
    const float* Wq   = (const float*)d_in[3];
    const float* Wk   = (const float*)d_in[4];
    const float* Wv   = (const float*)d_in[5];
    const float* Wo   = (const float*)d_in[6];
    const float* bo   = (const float*)d_in[7];
    const float* Wg   = (const float*)d_in[8];
    const float* bg   = (const float*)d_in[9];
    float* out = (float*)d_out;

    char* ws = (char*)d_ws;
    size_t off = 0;
    auto alloc = [&](size_t bytes) -> void* {
        void* p = ws + off;
        off += (bytes + 255) & ~(size_t)255;
        return p;
    };
    const size_t NX = (size_t)MROW * CIN;       // 8192*512
    const size_t NW = (size_t)NDIM * KDIM;      // 512*512
    const size_t NH = (size_t)Bn * H * Sq * Ch; // per-tensor head storage

    bf16* x_bf   = (bf16*)alloc(NX * 2);
    bf16* wq_bf  = (bf16*)alloc(NW * 2);
    bf16* wk_bf  = (bf16*)alloc(NW * 2);
    bf16* wv_bf  = (bf16*)alloc(NW * 2);
    bf16* wo_bf  = (bf16*)alloc(NW * 2);
    bf16* wg_bf  = (bf16*)alloc(NW * 2);
    bf16* q_bf   = (bf16*)alloc(NH * 2);
    bf16* k_bf   = (bf16*)alloc(NH * 2);
    bf16* vt_bf  = (bf16*)alloc(NH * 2);
    bf16* at_bf  = (bf16*)alloc(NX * 2);
    float* out_f = (float*)alloc(NX * 4);
    bf16* out_bf = (bf16*)alloc(NX * 2);

    cvt_f32_bf16<<<(int)((NX + 255) / 256), 256, 0, stream>>>(x,  x_bf,  (int)NX);
    cvt_f32_bf16<<<(int)((NW + 255) / 256), 256, 0, stream>>>(Wq, wq_bf, (int)NW);
    cvt_f32_bf16<<<(int)((NW + 255) / 256), 256, 0, stream>>>(Wk, wk_bf, (int)NW);
    cvt_f32_bf16<<<(int)((NW + 255) / 256), 256, 0, stream>>>(Wv, wv_bf, (int)NW);
    cvt_f32_bf16<<<(int)((NW + 255) / 256), 256, 0, stream>>>(Wo, wo_bf, (int)NW);
    cvt_f32_bf16<<<(int)((NW + 255) / 256), 256, 0, stream>>>(Wg, wg_bf, (int)NW);

    dim3 gg(MROW / 128, NDIM / 64);  // 64 x 8
    gemm512<<<gg, 256, 0, stream>>>(x_bf, wq_bf, nullptr, nullptr, q_bf,  nullptr, 0.125f, 0);
    gemm512<<<gg, 256, 0, stream>>>(x_bf, wk_bf, nullptr, nullptr, k_bf,  nullptr, 1.0f,   1);
    gemm512<<<gg, 256, 0, stream>>>(x_bf, wv_bf, nullptr, nullptr, vt_bf, nullptr, 1.0f,   2);

    dim3 ga(Sq / 128, Bn * H);       // 16 x 32
    attn_flash<<<ga, 256, 0, stream>>>(q_bf, k_bf, vt_bf, bias, am, at_bf);

    gemm512<<<gg, 256, 0, stream>>>(at_bf, wo_bf, bo, nullptr, out_bf, out_f, 1.0f, 3);
    gemm512<<<gg, 256, 0, stream>>>(out_bf, wg_bf, bg, out_f, nullptr, out, 1.0f, 4);
}
